// GTLayer_34961033790001
// MI455X (gfx1250) — matrix operand order repaired and checked by reading, not compiled, not run on hardware
//
#include <hip/hip_runtime.h>

#define N_NODES 4096
#define N_EDGES 262144
#define C_IN 5
#define C_OUT 2
#define NN (N_NODES * N_NODES)   // 16777216

typedef __attribute__((ext_vector_type(16))) __bf16        v16bf;
typedef __attribute__((ext_vector_type(8)))  float         v8f;
typedef __attribute__((ext_vector_type(8)))  unsigned int  v8u;
typedef __attribute__((ext_vector_type(4)))  int           i32x4;

#if defined(__gfx1250__) && __has_builtin(__builtin_amdgcn_global_load_async_to_lds_b128)
#define HAVE_ASYNC_LDS 1
#else
#define HAVE_ASYNC_LDS 0
#endif

__device__ __forceinline__ unsigned short f2bf(float f) {
    unsigned u = __builtin_bit_cast(unsigned, f);
    u += 0x7FFFu + ((u >> 16) & 1u);       // round-to-nearest-even
    return (unsigned short)(u >> 16);
}
__device__ __forceinline__ unsigned pk2(float a, float b) {
    return (unsigned)f2bf(a) | ((unsigned)f2bf(b) << 16);
}

// 16-byte global -> LDS copy: async path (ASYNCcnt-tracked) if available.
__device__ __forceinline__ void copy16_g2l(const unsigned short* g, unsigned short* l) {
#if HAVE_ASYNC_LDS
    typedef __attribute__((address_space(1))) i32x4 gi4;   // global int4
    typedef __attribute__((address_space(3))) i32x4 li4;   // LDS int4
    __builtin_amdgcn_global_load_async_to_lds_b128((gi4*)g, (li4*)l, 0, 0);
#else
    *reinterpret_cast<uint4*>(l) = *reinterpret_cast<const uint4*>(g);
#endif
}
__device__ __forceinline__ void wait_async_copies() {
#if HAVE_ASYNC_LDS
#if __has_builtin(__builtin_amdgcn_s_wait_asynccnt)
    __builtin_amdgcn_s_wait_asynccnt(0);
#else
    asm volatile("s_wait_asynccnt 0" ::: "memory");
#endif
#endif
}

// ---------------------------------------------------------------------------
// Softmax of weight1/weight2 rows (2x5 each) -> fbuf (ws) + f1/f2 output tail.
// ---------------------------------------------------------------------------
__global__ void softmax_rows(const float* __restrict__ w1,
                             const float* __restrict__ w2,
                             float* __restrict__ fbuf,
                             float* __restrict__ fout) {
    if (threadIdx.x == 0 && blockIdx.x == 0) {
        for (int s = 0; s < 2; ++s) {
            const float* w = s ? w2 : w1;
            for (int i = 0; i < C_OUT; ++i) {
                float m = w[i * C_IN];
                for (int c = 1; c < C_IN; ++c) m = fmaxf(m, w[i * C_IN + c]);
                float e[C_IN], sum = 0.f;
                for (int c = 0; c < C_IN; ++c) {
                    e[c] = __expf(w[i * C_IN + c] - m);
                    sum += e[c];
                }
                float inv = 1.f / sum;
                for (int c = 0; c < C_IN; ++c) {
                    float v = e[c] * inv;
                    fbuf[s * 10 + i * C_IN + c] = v;
                    fout[s * 10 + i * C_IN + c] = v;
                }
            }
        }
    }
}

// ---------------------------------------------------------------------------
// Zero the A/B accumulation buffers (grid-stride float4 stores).
// ---------------------------------------------------------------------------
__global__ void zero_f4(float4* __restrict__ p, long n4) {
    long i = blockIdx.x * (long)blockDim.x + threadIdx.x;
    long stride = gridDim.x * (long)blockDim.x;
    float4 z = make_float4(0.f, 0.f, 0.f, 0.f);
    for (; i < n4; i += stride) p[i] = z;
}

// ---------------------------------------------------------------------------
// Fused densify + channel mix via f32 global atomics.
// ---------------------------------------------------------------------------
__global__ void scatter_edges(const int*   __restrict__ eidx,
                              const float* __restrict__ eval_,
                              const float* __restrict__ fbuf,
                              float* __restrict__ Amat,
                              float* __restrict__ Bmat,
                              int oc) {
    int t = blockIdx.x * blockDim.x + threadIdx.x;   // < C_IN * N_EDGES
    int c = t >> 18;                                 // t / N_EDGES
    int e = t & (N_EDGES - 1);
    if (c >= C_IN) return;
    int   row = eidx[(c * 2 + 0) * N_EDGES + e];
    int   col = eidx[(c * 2 + 1) * N_EDGES + e];
    float v   = eval_[c * N_EDGES + e];
    float f1  = fbuf[oc * C_IN + c];
    float f2  = fbuf[10 + oc * C_IN + c];
    long  off = (long)row * N_NODES + col;
    atomicAdd(Amat + off, f1 * v);
    atomicAdd(Bmat + off, f2 * v);
}

// ---------------------------------------------------------------------------
// Elementwise f32 -> bf16 (row-major A operand).
// ---------------------------------------------------------------------------
__global__ void conv_bf16(const float4* __restrict__ src, uint2* __restrict__ dst, long n4) {
    long i = blockIdx.x * (long)blockDim.x + threadIdx.x;
    long stride = gridDim.x * (long)blockDim.x;
    for (; i < n4; i += stride) {
        float4 f = src[i];
        dst[i] = make_uint2(pk2(f.x, f.y), pk2(f.z, f.w));
    }
}

// ---------------------------------------------------------------------------
// B f32 [k][n] -> Bt bf16 [n][k]  (64x64 LDS-tiled transpose + convert).
// ---------------------------------------------------------------------------
__global__ __launch_bounds__(256)
void transpose_bf16(const float* __restrict__ B, unsigned short* __restrict__ Bt) {
    __shared__ float tile[64][65];
    const int t  = threadIdx.x;
    const int r  = t >> 2;           // 0..63
    const int c0 = (t & 3) * 16;     // 0,16,32,48
    const long kBase = (long)blockIdx.y * 64;
    const long nBase = (long)blockIdx.x * 64;

    const float4* sp = reinterpret_cast<const float4*>(B + (kBase + r) * N_NODES + nBase + c0);
    float4 v0 = sp[0], v1 = sp[1], v2 = sp[2], v3 = sp[3];
    float vv[16] = {v0.x, v0.y, v0.z, v0.w, v1.x, v1.y, v1.z, v1.w,
                    v2.x, v2.y, v2.z, v2.w, v3.x, v3.y, v3.z, v3.w};
    #pragma unroll
    for (int j = 0; j < 16; ++j) tile[r][c0 + j] = vv[j];
    __syncthreads();

    // write row n = nBase + r, ks kBase + c0 .. c0+15
    unsigned short* op = Bt + (nBase + r) * N_NODES + kBase + c0;
    uint4 o0 = make_uint4(pk2(tile[c0 + 0][r], tile[c0 + 1][r]),
                          pk2(tile[c0 + 2][r], tile[c0 + 3][r]),
                          pk2(tile[c0 + 4][r], tile[c0 + 5][r]),
                          pk2(tile[c0 + 6][r], tile[c0 + 7][r]));
    uint4 o1 = make_uint4(pk2(tile[c0 + 8][r], tile[c0 + 9][r]),
                          pk2(tile[c0 + 10][r], tile[c0 + 11][r]),
                          pk2(tile[c0 + 12][r], tile[c0 + 13][r]),
                          pk2(tile[c0 + 14][r], tile[c0 + 15][r]));
    *reinterpret_cast<uint4*>(op)     = o0;
    *reinterpret_cast<uint4*>(op + 8) = o1;
}

// ---------------------------------------------------------------------------
// H = A @ B with pre-formatted bf16 operands: Abf [m][k], Btbf [n][k].
// 256 threads (8 waves), 128x128 tile, K-step 64 (2 x 8 WMMA per stage).
// ---------------------------------------------------------------------------
#define BM 128
#define BN 128
#define BK 64
#define LDP 72   // LDS pitch in bf16 (144 B, 16B-aligned, bank stride 36)

__global__ __launch_bounds__(256, 2)
void gemm_bf16_wmma(const unsigned short* __restrict__ Abf,
                    const unsigned short* __restrict__ Btbf,
                    float* __restrict__ Hg) {
    __shared__ unsigned short As[BM * LDP];   // [row][k]
    __shared__ unsigned short Bs[BN * LDP];   // [n][k]

    const int t    = threadIdx.x;
    const int lane = t & 31;
    const int wave = t >> 5;
    const int wm   = wave >> 2;   // 0..1  -> 64-row strip
    const int wn   = wave & 3;    // 0..3  -> 32-col strip
    const int rowBase = blockIdx.y * BM;
    const int colBase = blockIdx.x * BN;

    // staging map: each thread copies 64 B of A and 64 B of B per K-step
    const int srow = t >> 1;          // 0..127 (row for A, n for B)
    const int skc  = (t & 1) * 32;    // k-chunk (bf16 elems)

    // fragment addressing (documented 16-bit WMMA layouts, wave32)
    const int lhalf = lane >> 4;
    const int nIdx  = lane & 15;
    const int klA   = lhalf * 8;
    const int klB   = lhalf * 8;

    v8f acc[4][2] = {};

    const unsigned short* ag = Abf  + (long)(rowBase + srow) * N_NODES + skc;
    const unsigned short* bg = Btbf + (long)(colBase + srow) * N_NODES + skc;
    unsigned short* al = &As[srow * LDP + skc];
    unsigned short* bl = &Bs[srow * LDP + skc];

    for (int k0 = 0; k0 < N_NODES; k0 += BK) {
        if (k0 + BK < N_NODES) {
            __builtin_prefetch((const char*)(ag + k0) + BK * 2, 0, 3);
            __builtin_prefetch((const char*)(bg + k0) + BK * 2, 0, 3);
        }
        __syncthreads();   // previous iteration's LDS reads complete
        #pragma unroll
        for (int j = 0; j < 4; ++j) {
            copy16_g2l(ag + k0 + j * 8, al + j * 8);
            copy16_g2l(bg + k0 + j * 8, bl + j * 8);
        }
        wait_async_copies();
        __syncthreads();

        #pragma unroll
        for (int kk = 0; kk < BK; kk += 32) {
            v16bf afr[4];
            #pragma unroll
            for (int mi = 0; mi < 4; ++mi) {
                const unsigned short* arp =
                    &As[(wm * 64 + mi * 16 + nIdx) * LDP + kk + klA];
                uint4 x = *reinterpret_cast<const uint4*>(arp);
                uint4 y = *reinterpret_cast<const uint4*>(arp + 16);
                v8u av = {x.x, x.y, x.z, x.w, y.x, y.y, y.z, y.w};
                afr[mi] = __builtin_bit_cast(v16bf, av);
            }
            v16bf bfr[2];
            #pragma unroll
            for (int ni = 0; ni < 2; ++ni) {
                const unsigned short* brp =
                    &Bs[(wn * 32 + ni * 16 + nIdx) * LDP + kk + klB];
                uint4 x = *reinterpret_cast<const uint4*>(brp);
                uint4 y = *reinterpret_cast<const uint4*>(brp + 16);
                v8u bv8 = {x.x, x.y, x.z, x.w, y.x, y.y, y.z, y.w};
                bfr[ni] = __builtin_bit_cast(v16bf, bv8);
            }
            #pragma unroll
            for (int mi = 0; mi < 4; ++mi)
                #pragma unroll
                for (int ni = 0; ni < 2; ++ni)
                    acc[mi][ni] = __builtin_amdgcn_wmma_f32_16x16x32_bf16(
                        false, afr[mi], false, bfr[ni],
                        (short)0, acc[mi][ni], false, false);
        }
    }

    #pragma unroll
    for (int mi = 0; mi < 4; ++mi) {
        #pragma unroll
        for (int ni = 0; ni < 2; ++ni) {
            int r0 = rowBase + wm * 64 + mi * 16 + lhalf * 8;
            int cc = colBase + wn * 32 + ni * 16 + nIdx;
            #pragma unroll
            for (int r = 0; r < 8; ++r)
                Hg[(long)(r0 + r) * N_NODES + cc] = acc[mi][ni][r];
        }
    }
}

// ---------------------------------------------------------------------------
// WS: A f32 [NN] | B f32 [NN] | Abf bf16 [NN] | Btbf bf16 [NN] | fbuf (201 MB)
// Per output channel (serialized): zero -> scatter -> convert -> GEMM.
// ---------------------------------------------------------------------------
extern "C" void kernel_launch(void* const* d_in, const int* in_sizes, int n_in,
                              void* d_out, int out_size, void* d_ws, size_t ws_size,
                              hipStream_t stream) {
    const int*   edge_index = (const int*)d_in[0];
    const float* edge_value = (const float*)d_in[1];
    const float* w1         = (const float*)d_in[2];
    const float* w2         = (const float*)d_in[3];

    float* H    = (float*)d_out;
    float* fout = H + (long)C_OUT * NN;        // f1(10) ++ f2(10) tail

    float*          Amat = (float*)d_ws;
    float*          Bmat = Amat + (long)NN;
    unsigned short* Abf  = (unsigned short*)(Bmat + (long)NN);
    unsigned short* Btbf = Abf + (long)NN;
    float*          fbuf = (float*)(Btbf + (long)NN);

    softmax_rows<<<1, 32, 0, stream>>>(w1, w2, fbuf, fout);

    for (int oc = 0; oc < C_OUT; ++oc) {
        zero_f4<<<4096, 256, 0, stream>>>((float4*)Amat, (long)(2 * (long)NN) / 4);
        scatter_edges<<<(C_IN * N_EDGES) / 256, 256, 0, stream>>>(
            edge_index, edge_value, fbuf, Amat, Bmat, oc);
        conv_bf16<<<4096, 256, 0, stream>>>((const float4*)Amat, (uint2*)Abf, (long)NN / 4);
        transpose_bf16<<<dim3(64, 64), 256, 0, stream>>>(Bmat, Btbf);
        dim3 g(N_NODES / BN, N_NODES / BM);
        gemm_bf16_wmma<<<g, 256, 0, stream>>>(Abf, Btbf, H + (long)oc * NN);
    }
}
